// DCGAN_G_2774548873929
// MI455X (gfx1250) — matrix operand order repaired and checked by reading, not compiled, not run on hardware
//
#include <hip/hip_runtime.h>
#include <hip/hip_bf16.h>

#define BB 32
#define NN 128
#define KK 48
#define FF 256
#define GG 50
#define TT 3
#define PI_F 3.14159265358979f
#define CUT 5.0f

typedef __attribute__((ext_vector_type(16))) _Float16 v16h;
typedef __attribute__((ext_vector_type(8)))  _Float16 v8h;
typedef __attribute__((ext_vector_type(8)))  float    v8f;

typedef unsigned int u32x4 __attribute__((ext_vector_type(4)));
typedef int          i32x4 __attribute__((ext_vector_type(4)));
typedef int          i32x8 __attribute__((ext_vector_type(8)));

#if __has_builtin(__builtin_amdgcn_tensor_load_to_lds) && \
    __has_builtin(__builtin_amdgcn_s_wait_tensorcnt)
#define USE_TDM 1
#else
#define USE_TDM 0
#endif

__device__ __forceinline__ v8f zerov() {
    v8f z = {0.f,0.f,0.f,0.f,0.f,0.f,0.f,0.f};
    return z;
}

__device__ __forceinline__ float sspf(float x) {
    // softplus(x) - ln2, numerically stable
    float ax = fabsf(x);
    return fmaxf(x, 0.f) + log1pf(__expf(-ax)) - 0.69314718055994531f;
}

__device__ __forceinline__ float sigmf(float x) {
    return 1.f / (1.f + __expf(-x));
}

__device__ __forceinline__ v16h mk16(v8h lo, v8h hi) {
    return __builtin_shufflevector(lo, hi, 0,1,2,3,4,5,6,7,8,9,10,11,12,13,14,15);
}

// A fragment (16x32 f16): lane 0-15 -> M=lane, K in {kb..kb+15},
// kb = 16*(lane>=16). Row-major source [rows, lda].
__device__ __forceinline__ v16h loadAfrag(const _Float16* base, int lda,
                                          int rowBase, int k0, int lane) {
    int m  = lane & 15;
    int kb = (lane >> 4) * 16;
    const _Float16* p = base + (size_t)(rowBase + m) * lda + k0 + kb;
    v8h lo = *(const v8h*)p;
    v8h hi = *(const v8h*)(p + 8);
    return mk16(lo, hi);
}

// B fragment (32x16 f16) from N-major (transposed) weights WT[N][ldk]:
// lane holds column n = n0+(lane&15), K = k0 + 16*(lane>=16) + j, j=0..15 contiguous.
__device__ __forceinline__ v16h loadBfrag(const _Float16* wT, int ldk,
                                          int n0, int k0, int lane) {
    int n  = n0 + (lane & 15);
    int kb = k0 + (lane >> 4) * 16;
    return *(const v16h*)(wT + (size_t)n * ldk + kb);
}

__device__ __forceinline__ v8f wmma_f16(v16h a, v16h b, v8f c) {
    return __builtin_amdgcn_wmma_f32_16x16x32_f16(false, a, false, b,
                                                  (short)0, c, false, false);
}

#if USE_TDM
// Issue one TDM 2-D tile load: tile_d1 rows x tile_d0 contiguous f16 elements,
// row stride = stride0 elements, destination LDS at lds_addr (bytes).
// D# packing per CDNA5 ISA 8.3/8.4 (count=1, type=2 "image", data_size=2B).
// Toolchain uses the 6-arg builtin: (g0 u32x4, g1 i32x8, i32x4, i32x4, i32x8, cpol).
__device__ __forceinline__ void tdm_load_2d_f16(unsigned lds_addr,
                                                const void* gtile,
                                                unsigned tile_d0, unsigned tile_d1,
                                                unsigned long long stride0) {
    unsigned long long ga = (unsigned long long)gtile;
    u32x4 g0;
    g0[0] = 1u;                                         // count=1, user desc
    g0[1] = lds_addr;                                   // lds_addr [63:32]
    g0[2] = (unsigned)(ga & 0xffffffffu);               // global_addr lo
    g0[3] = (unsigned)((ga >> 32) & 0x01ffffffu)        // global_addr hi (57b)
          | 0x80000000u;                                // type=2 -> bits 127:126
    i32x8 g1;
    g1[0] = 0x00010000;                                 // data_size=1 (2 bytes)
    g1[1] = (int)((tile_d0 & 0xffffu) << 16);           // tensor_dim0[15:0]
    g1[2] = (int)(((tile_d0 >> 16) & 0xffffu)           // tensor_dim0[31:16]
          | ((tile_d1 & 0xffffu) << 16));               // tensor_dim1[15:0]
    g1[3] = (int)(((tile_d1 >> 16) & 0xffffu)           // tensor_dim1[31:16]
          | ((tile_d0 & 0xffffu) << 16));               // tile_dim0
    g1[4] = (int)(tile_d1 & 0xffffu);                   // tile_dim1 (tile_dim2=0)
    g1[5] = (int)(unsigned)(stride0 & 0xffffffffu);     // tensor_dim0_stride lo
    g1[6] = (int)(unsigned)((stride0 >> 32) & 0xffffu); // stride hi (dim1_stride=0)
    g1[7] = 0;
    i32x4 gz; gz[0] = gz[1] = gz[2] = gz[3] = 0;        // groups 2/3 unused (2-D)
    i32x8 gz8;
    gz8[0]=gz8[1]=gz8[2]=gz8[3]=gz8[4]=gz8[5]=gz8[6]=gz8[7]=0;
    __builtin_amdgcn_tensor_load_to_lds(g0, g1, gz, gz, gz8, 0);
}
#endif

// ---------------------------------------------------------------------------
// Weight transpose+convert: src f32 [Ksrc, N] row-major -> dst f16 [N, Kpad]
// ---------------------------------------------------------------------------
__global__ __launch_bounds__(256) void convT_kernel(const float* __restrict__ src,
                                                    _Float16* __restrict__ dst,
                                                    int Ksrc, int Kpad, int N) {
    int i = blockIdx.x * 256 + threadIdx.x;
    if (i >= N * Kpad) return;
    int n = i / Kpad, k = i - n * Kpad;
    dst[i] = (k < Ksrc) ? (_Float16)src[(size_t)k * N + n] : (_Float16)0.f;
}

// ---------------------------------------------------------------------------
// Embedding gather: x32/x16 [B*N, F]
// ---------------------------------------------------------------------------
__global__ __launch_bounds__(256) void embed_kernel(const int* __restrict__ zn,
                                                    const float* __restrict__ emb,
                                                    float* __restrict__ x32,
                                                    _Float16* __restrict__ x16) {
    int bn = blockIdx.x, f = threadIdx.x;
    int z = zn[bn];
    float v = emb[(size_t)z * FF + f];
    x32[(size_t)bn * FF + f] = v;
    x16[(size_t)bn * FF + f] = (_Float16)v;
}

// ---------------------------------------------------------------------------
// Generic WMMA GEMM: out = [resid +] act(A(f16)[M,K] @ WT(f16)[N,K]^T + bias)
// block = 256 threads (8 waves); wave = 16 rows x 64 cols; block = 128 x 64.
// Weight slab (64 x 32 f16 per K-step) staged to LDS by the Tensor Data Mover,
// double-buffered so the next DMA overlaps the current WMMA work.
// ---------------------------------------------------------------------------
__global__ __launch_bounds__(256) void gemm_kernel(
    const _Float16* __restrict__ A, const _Float16* __restrict__ WT,
    const float* __restrict__ bias, float* __restrict__ out32,
    _Float16* __restrict__ out16, const float* __restrict__ resid,
    int M, int K, int N, int act)
{
    int lane = threadIdx.x & 31, wv = threadIdx.x >> 5;
    int nTiles = N >> 6;
    int cb = blockIdx.x % nTiles, rb = blockIdx.x / nTiles;
    int rowBase = rb * 128 + wv * 16;
    int n0 = cb * 64;

    v8f acc[4];
#pragma unroll
    for (int i = 0; i < 4; i++) acc[i] = zerov();

    int nl = lane & 15, mh = (lane >> 4) * 8;

#if USE_TDM
    __shared__ _Float16 sB[2][64 * 32];   // double-buffered weight slab
    const int S = K >> 5;
    if (threadIdx.x == 0) {
        tdm_load_2d_f16((unsigned)(uintptr_t)&sB[0][0],
                        WT + (size_t)n0 * K, 32u, 64u,
                        (unsigned long long)K);
    }
    for (int i = 0; i < S; i++) {
        if (threadIdx.x == 0) {
            if (i + 1 < S) {
                tdm_load_2d_f16((unsigned)(uintptr_t)&sB[(i + 1) & 1][0],
                                WT + (size_t)n0 * K + (i + 1) * 32, 32u, 64u,
                                (unsigned long long)K);
                __builtin_amdgcn_s_wait_tensorcnt(1);   // current slab landed
            } else {
                __builtin_amdgcn_s_wait_tensorcnt(0);
            }
        }
        __syncthreads();                                 // slab visible to WGP
        const _Float16* bs = &sB[i & 1][0];
        v16h a = loadAfrag(A, K, rowBase, i * 32, lane);
        int kb = (lane >> 4) * 16;
#pragma unroll
        for (int ct = 0; ct < 4; ct++) {
            v16h b = *(const v16h*)(bs + (ct * 16 + nl) * 32 + kb);
            acc[ct] = wmma_f16(a, b, acc[ct]);
        }
        __syncthreads();                                 // reads done before reuse
    }
#else
    for (int k0 = 0; k0 < K; k0 += 32) {
        if (k0 + 32 < K)
            __builtin_prefetch(WT + (size_t)(n0 + nl) * K + k0 + 32, 0, 1);
        v16h a = loadAfrag(A, K, rowBase, k0, lane);
#pragma unroll
        for (int ct = 0; ct < 4; ct++) {
            v16h b = loadBfrag(WT, K, n0 + ct * 16, k0, lane);
            acc[ct] = wmma_f16(a, b, acc[ct]);
        }
    }
#endif

#pragma unroll
    for (int ct = 0; ct < 4; ct++) {
        int n = n0 + ct * 16 + nl;
        float bv = bias ? bias[n] : 0.f;
#pragma unroll
        for (int v = 0; v < 8; v++) {
            int m = rowBase + mh + v;
            float val = acc[ct][v] + bv;
            if (act) val = sspf(val);
            if (resid) val += resid[(size_t)m * N + n];
            if (out32) out32[(size_t)m * N + n] = val;
            if (out16) out16[(size_t)m * N + n] = (_Float16)val;
        }
    }
}

// ---------------------------------------------------------------------------
// Fused continuous-filter convolution: one block per (b,n).
//   d/fcut/fexp -> LDS; h1 = ssp(fexp@W1+b1) (WMMA, LDS); W = (h1@W2+b2)*fcut
//   (WMMA, LDS); agg[f] = sum_k W[k,f] * y[neighbor(k), f]  -> agg16
// ---------------------------------------------------------------------------
__global__ __launch_bounds__(256) void fused_cfconv_kernel(
    const float* __restrict__ positions, const int* __restrict__ neighbors,
    const float* __restrict__ nmask,
    const _Float16* __restrict__ wT1, const float* __restrict__ b1,
    const _Float16* __restrict__ wT2, const float* __restrict__ b2,
    const float* __restrict__ y, _Float16* __restrict__ agg16)
{
    __shared__ _Float16 sA1[KK * 72];    // fexp tile, padded stride
    __shared__ _Float16 sH1[KK * 264];   // hidden activations
    __shared__ _Float16 sW [KK * 264];   // filters * fcut
    __shared__ float sD[KK], sFc[KK];
    __shared__ int   sNb[KK];

    int bn = blockIdx.x;        // b*N + n
    int b  = bn >> 7;           // / NN
    int tid = threadIdx.x, lane = tid & 31, wv = tid >> 5;

    // geometry
    if (tid < KK) {
        float px = positions[(size_t)bn * 3 + 0];
        float py = positions[(size_t)bn * 3 + 1];
        float pz = positions[(size_t)bn * 3 + 2];
        int nb = neighbors[(size_t)bn * KK + tid];
        const float* pj = positions + (size_t)(b * NN + nb) * 3;
        float dx = pj[0] - px, dy = pj[1] - py, dz = pj[2] - pz;
        float d = sqrtf(dx * dx + dy * dy + dz * dz + 1e-8f);
        float fc = 0.5f * (__cosf(PI_F * d / CUT) + 1.f);
        fc *= (d < CUT) ? 1.f : 0.f;
        fc *= nmask[(size_t)bn * KK + tid];
        sD[tid] = d; sFc[tid] = fc; sNb[tid] = nb;
    }
    __syncthreads();

    // gaussian expansion -> sA1 (f16, K padded 50 -> 64)
    const float delta = CUT / (GG - 1);
    const float coeff = -0.5f / (delta * delta);
    for (int i = tid; i < KK * 64; i += 256) {
        int k = i >> 6, g = i & 63;
        float val = 0.f;
        if (g < GG) { float dd = sD[k] - g * delta; val = __expf(coeff * dd * dd); }
        sA1[k * 72 + g] = (_Float16)val;
    }
    __syncthreads();

    int n0 = wv * 32;                 // wave's 32-col slab of F=256
    int nl = lane & 15, mh = (lane >> 4) * 8;

    // ---- GEMM1: 48x64x256, ssp epilogue -> sH1 ----
    {
        v8f acc[3][2];
#pragma unroll
        for (int r = 0; r < 3; r++) for (int c = 0; c < 2; c++) acc[r][c] = zerov();
#pragma unroll
        for (int k0 = 0; k0 < 64; k0 += 32) {
            v16h a0 = loadAfrag(sA1, 72,  0, k0, lane);
            v16h a1 = loadAfrag(sA1, 72, 16, k0, lane);
            v16h a2 = loadAfrag(sA1, 72, 32, k0, lane);
#pragma unroll
            for (int ct = 0; ct < 2; ct++) {
                v16h bf = loadBfrag(wT1, 64, n0 + ct * 16, k0, lane);
                acc[0][ct] = wmma_f16(a0, bf, acc[0][ct]);
                acc[1][ct] = wmma_f16(a1, bf, acc[1][ct]);
                acc[2][ct] = wmma_f16(a2, bf, acc[2][ct]);
            }
        }
#pragma unroll
        for (int rt = 0; rt < 3; rt++)
#pragma unroll
            for (int ct = 0; ct < 2; ct++) {
                int n = n0 + ct * 16 + nl;
                float bv = b1[n];
#pragma unroll
                for (int v = 0; v < 8; v++) {
                    int m = rt * 16 + mh + v;
                    sH1[m * 264 + n] = (_Float16)sspf(acc[rt][ct][v] + bv);
                }
            }
    }
    __syncthreads();

    // ---- GEMM2: 48x256x256, (+b2)*fcut epilogue -> sW ----
    {
        v8f acc[3][2];
#pragma unroll
        for (int r = 0; r < 3; r++) for (int c = 0; c < 2; c++) acc[r][c] = zerov();
        for (int k0 = 0; k0 < 256; k0 += 32) {
            if (k0 + 32 < 256)
                __builtin_prefetch(wT2 + (size_t)(n0 + nl) * 256 + k0 + 32, 0, 1);
            v16h a0 = loadAfrag(sH1, 264,  0, k0, lane);
            v16h a1 = loadAfrag(sH1, 264, 16, k0, lane);
            v16h a2 = loadAfrag(sH1, 264, 32, k0, lane);
#pragma unroll
            for (int ct = 0; ct < 2; ct++) {
                v16h bf = loadBfrag(wT2, 256, n0 + ct * 16, k0, lane);
                acc[0][ct] = wmma_f16(a0, bf, acc[0][ct]);
                acc[1][ct] = wmma_f16(a1, bf, acc[1][ct]);
                acc[2][ct] = wmma_f16(a2, bf, acc[2][ct]);
            }
        }
#pragma unroll
        for (int rt = 0; rt < 3; rt++)
#pragma unroll
            for (int ct = 0; ct < 2; ct++) {
                int n = n0 + ct * 16 + nl;
                float bv = b2[n];
#pragma unroll
                for (int v = 0; v < 8; v++) {
                    int m = rt * 16 + mh + v;
                    sW[m * 264 + n] = (_Float16)((acc[rt][ct][v] + bv) * sFc[m]);
                }
            }
    }
    __syncthreads();

    // ---- neighbor gather-reduce: agg[f] = sum_k W[k,f] * y[nb(k), f] ----
    {
        int f = tid;
        float s = 0.f;
        size_t ybase = (size_t)b * NN * FF;
#pragma unroll 4
        for (int k = 0; k < KK; k++)
            s += (float)sW[k * 264 + f] * y[ybase + (size_t)sNb[k] * FF + f];
        agg16[(size_t)bn * FF + f] = (_Float16)s;
    }
}

// ---------------------------------------------------------------------------
// Head: displacement, fractional coords, predictor MLP, masked mean,
// new_cell + new_pos. One block per structure b.
// ---------------------------------------------------------------------------
__global__ __launch_bounds__(128) void head_kernel(
    const float* __restrict__ xf, const float* __restrict__ positions,
    const float* __restrict__ cell, const float* __restrict__ amask,
    const float* __restrict__ fd_w, const float* __restrict__ fd_b,
    const float* __restrict__ pw1, const float* __restrict__ pb1,
    const float* __restrict__ pw2, const float* __restrict__ pb2,
    float* __restrict__ out)
{
    __shared__ float sInv[9], sAcc[10], sNC[9];
    int b = blockIdx.x, n = threadIdx.x;

    if (n == 0) {
        const float* c = cell + (size_t)b * 9;
        float a00=c[0],a01=c[1],a02=c[2],a10=c[3],a11=c[4],a12=c[5],a20=c[6],a21=c[7],a22=c[8];
        float det = a00*(a11*a22-a12*a21) - a01*(a10*a22-a12*a20) + a02*(a10*a21-a11*a20);
        float id = 1.f / det;
        sInv[0]=(a11*a22-a12*a21)*id; sInv[1]=(a02*a21-a01*a22)*id; sInv[2]=(a01*a12-a02*a11)*id;
        sInv[3]=(a12*a20-a10*a22)*id; sInv[4]=(a00*a22-a02*a20)*id; sInv[5]=(a02*a10-a00*a12)*id;
        sInv[6]=(a10*a21-a11*a20)*id; sInv[7]=(a01*a20-a00*a21)*id; sInv[8]=(a00*a11-a01*a10)*id;
    }
    if (n < 10) sAcc[n] = 0.f;
    __syncthreads();

    // displacement from features
    const float* xr = xf + ((size_t)b * NN + n) * FF;
    float u0 = fd_b[0], u1 = fd_b[1], u2 = fd_b[2];
    for (int f = 0; f < FF; f++) {
        float xv = xr[f];
        u0 += xv * fd_w[f * 3 + 0];
        u1 += xv * fd_w[f * 3 + 1];
        u2 += xv * fd_w[f * 3 + 2];
    }
    float dsp[3] = { sigmf(10.f * u0), sigmf(10.f * u1), sigmf(10.f * u2) };

    const float* p = positions + ((size_t)b * NN + n) * 3;
    float fr[3];
#pragma unroll
    for (int c = 0; c < 3; c++) {
        float v = p[0] * sInv[0 * 3 + c] + p[1] * sInv[1 * 3 + c] + p[2] * sInv[2 * 3 + c] + dsp[c];
        fr[c] = v - floorf(v);
    }
    const float* cl = cell + (size_t)b * 9;
    float rp[3];
#pragma unroll
    for (int c = 0; c < 3; c++)
        rp[c] = fr[0] * cl[0 * 3 + c] + fr[1] * cl[1 * 3 + c] + fr[2] * cl[2 * 3 + c];

    float h1[6];
#pragma unroll
    for (int j = 0; j < 6; j++)
        h1[j] = sspf(rp[0] * pw1[0 * 6 + j] + rp[1] * pw1[1 * 6 + j] + rp[2] * pw1[2 * 6 + j] + pb1[j]);

    float msk = amask[(size_t)b * NN + n];
#pragma unroll
    for (int o = 0; o < 9; o++) {
        float h2 = pb2[o];
#pragma unroll
        for (int j = 0; j < 6; j++) h2 += h1[j] * pw2[j * 9 + o];
        atomicAdd(&sAcc[o], h2 * msk);
    }
    atomicAdd(&sAcc[9], msk);
    __syncthreads();

    if (n == 0) {
        float num = sAcc[9];
        float scale = 3.f * cbrtf(num);
        for (int o = 0; o < 9; o++) {
            float eye = (o == 0 || o == 4 || o == 8) ? 1.f : 0.f;
            float ncv = scale * (eye + sAcc[o] / num);
            out[(size_t)BB * NN * 3 + (size_t)b * 9 + o] = ncv;
            sNC[o] = ncv;
        }
    }
    __syncthreads();

#pragma unroll
    for (int c = 0; c < 3; c++)
        out[((size_t)b * NN + n) * 3 + c] =
            fr[0] * sNC[0 * 3 + c] + fr[1] * sNC[1 * 3 + c] + fr[2] * sNC[2 * 3 + c];
}

// ---------------------------------------------------------------------------
extern "C" void kernel_launch(void* const* d_in, const int* in_sizes, int n_in,
                              void* d_out, int out_size, void* d_ws, size_t ws_size,
                              hipStream_t stream) {
    (void)in_sizes; (void)n_in; (void)out_size; (void)ws_size;
    const float* positions = (const float*)d_in[0];
    const float* cell      = (const float*)d_in[1];
    const int*   zn        = (const int*)d_in[2];
    const int*   neighbors = (const int*)d_in[3];
    const float* nmask     = (const float*)d_in[4];
    const float* amask     = (const float*)d_in[5];
    const float* embedding = (const float*)d_in[6];
    const float* filt_w1   = (const float*)d_in[7];
    const float* filt_b1   = (const float*)d_in[8];
    const float* filt_w2   = (const float*)d_in[9];
    const float* filt_b2   = (const float*)d_in[10];
    const float* in2f_w    = (const float*)d_in[11];
    const float* f2out_w1  = (const float*)d_in[12];
    const float* f2out_b1  = (const float*)d_in[13];
    const float* f2out_w2  = (const float*)d_in[14];
    const float* f2out_b2  = (const float*)d_in[15];
    const float* fd_w      = (const float*)d_in[16];
    const float* fd_b      = (const float*)d_in[17];
    const float* pw1       = (const float*)d_in[18];
    const float* pb1       = (const float*)d_in[19];
    const float* pw2       = (const float*)d_in[20];
    const float* pb2       = (const float*)d_in[21];

    char* ws = (char*)d_ws;
    size_t off = 0;
    auto alloc = [&](size_t bytes) -> char* {
        char* pp = ws + off;
        off += (bytes + 255) & ~(size_t)255;
        return pp;
    };
    const size_t NODES = (size_t)BB * NN;              // 4096
    float*    X32  = (float*)   alloc(NODES * FF * 4);
    _Float16* X16  = (_Float16*)alloc(NODES * FF * 2);
    float*    Y    = (float*)   alloc(NODES * FF * 4);
    _Float16* AGG  = (_Float16*)alloc(NODES * FF * 2);
    _Float16* P16  = (_Float16*)alloc(NODES * FF * 2);
    _Float16* WT1  = (_Float16*)alloc((size_t)TT * FF * 64 * 2);
    _Float16* WT2  = (_Float16*)alloc((size_t)TT * FF * FF * 2);
    _Float16* IN2FT= (_Float16*)alloc((size_t)TT * FF * FF * 2);
    _Float16* FO1T = (_Float16*)alloc((size_t)TT * FF * FF * 2);
    _Float16* FO2T = (_Float16*)alloc((size_t)TT * FF * FF * 2);

    // weight prep (transpose + f16 convert, pad G 50->64)
    for (int t = 0; t < TT; t++) {
        convT_kernel<<<(FF * 64 + 255) / 256, 256, 0, stream>>>(
            filt_w1 + (size_t)t * GG * FF, WT1 + (size_t)t * FF * 64, GG, 64, FF);
        convT_kernel<<<(FF * FF + 255) / 256, 256, 0, stream>>>(
            filt_w2 + (size_t)t * FF * FF, WT2 + (size_t)t * FF * FF, FF, FF, FF);
        convT_kernel<<<(FF * FF + 255) / 256, 256, 0, stream>>>(
            in2f_w + (size_t)t * FF * FF, IN2FT + (size_t)t * FF * FF, FF, FF, FF);
        convT_kernel<<<(FF * FF + 255) / 256, 256, 0, stream>>>(
            f2out_w1 + (size_t)t * FF * FF, FO1T + (size_t)t * FF * FF, FF, FF, FF);
        convT_kernel<<<(FF * FF + 255) / 256, 256, 0, stream>>>(
            f2out_w2 + (size_t)t * FF * FF, FO2T + (size_t)t * FF * FF, FF, FF, FF);
    }

    embed_kernel<<<NODES, 256, 0, stream>>>(zn, embedding, X32, X16);

    const int gemmGrid = (4096 / 128) * (FF / 64);  // 128 blocks
    for (int t = 0; t < TT; t++) {
        // y = x @ in2f_w[t]
        gemm_kernel<<<gemmGrid, 256, 0, stream>>>(
            X16, IN2FT + (size_t)t * FF * FF, nullptr, Y, nullptr, nullptr,
            4096, FF, FF, 0);
        // fused filter-net + cutoff + gather-reduce -> agg
        fused_cfconv_kernel<<<NODES, 256, 0, stream>>>(
            positions, neighbors, nmask,
            WT1 + (size_t)t * FF * 64, filt_b1 + (size_t)t * FF,
            WT2 + (size_t)t * FF * FF, filt_b2 + (size_t)t * FF,
            Y, AGG);
        // post1 = ssp(agg @ f2out_w1 + b1)
        gemm_kernel<<<gemmGrid, 256, 0, stream>>>(
            AGG, FO1T + (size_t)t * FF * FF, f2out_b1 + (size_t)t * FF,
            nullptr, P16, nullptr, 4096, FF, FF, 1);
        // x = x + post1 @ f2out_w2 + b2   (updates both f32 and f16 copies)
        gemm_kernel<<<gemmGrid, 256, 0, stream>>>(
            P16, FO2T + (size_t)t * FF * FF, f2out_b2 + (size_t)t * FF,
            X32, X16, X32, 4096, FF, FF, 0);
    }

    head_kernel<<<BB, NN, 0, stream>>>(X32, positions, cell, amask,
                                       fd_w, fd_b, pw1, pb1, pw2, pb2,
                                       (float*)d_out);
}
